// LatticeQuantizedParameter_75642964017924
// MI455X (gfx1250) — matrix operand order repaired and checked by reading, not compiled, not run on hardware
//
#include <hip/hip_runtime.h>

typedef __attribute__((ext_vector_type(16))) _Float16 v16h;
typedef __attribute__((ext_vector_type(8)))  _Float16 v8h;
typedef __attribute__((ext_vector_type(8)))  float    v8f;
typedef __attribute__((ext_vector_type(4)))  float    v4f;

#define TOKN   4096
#define NDIM   8192
#define MDIM   8192
#define SCALEV 0.9f

#define BT 128          // token tile per block
#define BM 128          // out-feature tile per block
#define KT 64           // K per main-loop iteration (two WMMA k-phases)
#define LP 72           // LDS row pitch in halves (144 B: 16B-aligned, conflict-free)
#define NSTEPS (NDIM / KT)

static __device__ __forceinline__ v16h cat16(v8h lo, v8h hi) {
  v16h r;
#pragma unroll
  for (int i = 0; i < 8; ++i) { r[i] = lo[i]; r[i + 8] = hi[i]; }
  return r;
}

static __device__ __forceinline__ v8h cvt8(v4f a, v4f b) {
  v8h h;
  h[0] = (_Float16)a[0]; h[1] = (_Float16)a[1];
  h[2] = (_Float16)a[2]; h[3] = (_Float16)a[3];
  h[4] = (_Float16)b[0]; h[5] = (_Float16)b[1];
  h[6] = (_Float16)b[2]; h[7] = (_Float16)b[3];
  return h;
}

// ---- prepass: x fp32 -> fp16 (written once into workspace; L2-resident after) ----
__global__ __launch_bounds__(256)
void xcvt_kernel(const float* __restrict__ x, _Float16* __restrict__ xh) {
  size_t i = ((size_t)blockIdx.x * 256 + threadIdx.x) * 8;
  v4f a = *(const v4f*)(x + i);
  v4f b = *(const v4f*)(x + i + 4);
  *(v8h*)(xh + i) = cvt8(a, b);
}

// ---- main GEMM: decode-on-the-fly E8P weights, f16 WMMA, double-buffered LDS,
// ----           async global->LDS staging of the activation tile (ASYNCcnt) ----
template <bool XF16>
__global__ __launch_bounds__(256)
void e8p_gemm_kernel(const void* __restrict__ xin,
                     const int*  __restrict__ idxs,
                     const float* __restrict__ grid,
                     float* __restrict__ out)
{
  __shared__ _Float16 lx[2][BT * LP];   // 2 x 18 KB: x tile (f16)
  __shared__ _Float16 lw[2][BM * LP];   // 2 x 18 KB: decoded W tile (f16)

  const int tid  = threadIdx.x;
  const int lane = tid & 31;
  const int wave = tid >> 5;
  const int wy   = wave >> 1;    // token group 0..3 (32 tokens each)
  const int wx   = wave & 1;     // feature group 0..1 (64 feats each)

  const int t0 = blockIdx.y * BT;
  const int m0 = blockIdx.x * BM;

  const int lhalf = lane >> 4;
  const int llow  = lane & 15;

  // x staging coords: each thread owns 32 halves (64 B) of one row
  const int xrow = tid >> 1;          // 0..127
  const int xs   = (tid & 1) * 32;    // 0 or 32 (halves)

  v8f acc[2][4] = {};

  auto stage = [&](int ks, int buf) {
    const int k0 = ks * KT;
    _Float16* dx = &lx[buf][0];
    _Float16* dw = &lw[buf][0];

    // x tile: BT x KT halves
    if (XF16) {
      // async DMA: 4 x 16 B per lane, memory -> LDS, no VGPR round-trip.
      // Same 64 B are contiguous on both sides, so one base pair + imm offsets.
      const _Float16* src = (const _Float16*)xin + (size_t)(t0 + xrow) * NDIM + k0 + xs;
      unsigned ldsa = (unsigned)(size_t)(dx + xrow * LP + xs);
      asm volatile(
          "global_load_async_to_lds_b128 %0, %1, off\n\t"
          "global_load_async_to_lds_b128 %0, %1, off offset:16\n\t"
          "global_load_async_to_lds_b128 %0, %1, off offset:32\n\t"
          "global_load_async_to_lds_b128 %0, %1, off offset:48"
          :: "v"(ldsa), "v"(src) : "memory");
    } else {
      const float* src = (const float*)xin + (size_t)(t0 + xrow) * NDIM + k0 + xs;
#pragma unroll
      for (int j = 0; j < 4; ++j) {
        v4f a = *(const v4f*)(src + j * 8);
        v4f b = *(const v4f*)(src + j * 8 + 4);
        *(v8h*)(dx + xrow * LP + xs + j * 8) = cvt8(a, b);
      }
    }

    // W tile: BM x KT halves from 1024 codes (4 per thread), codebook gather
#pragma unroll
    for (int it = 0; it < 4; ++it) {
      int i  = tid + it * 256;   // 0..1023
      int mr = i >> 3;           // 0..127
      int cg = i & 7;            // k = cg*8
      int code = idxs[(size_t)(m0 + mr) * (NDIM / 8) + (k0 >> 3) + cg];
      const float* gp = grid + (size_t)code * 8;
      *(v8h*)(dw + mr * LP + cg * 8) = cvt8(*(const v4f*)gp, *(const v4f*)(gp + 4));
    }
  };

  stage(0, 0);

  for (int ks = 0; ks < NSTEPS; ++ks) {
    const int buf = ks & 1;
    // all of this thread's async LDS writes for `buf` must land before the
    // barrier that publishes the buffer to the other waves
    if (XF16) asm volatile("s_wait_asynccnt 0" ::: "memory");
    __syncthreads();

    if (ks + 1 < NSTEPS) {
      if (ks + 2 < NSTEPS) {
        // pull x for step ks+2 toward L2 (global_prefetch_b8)
        const char* pf;
        if (XF16) pf = (const char*)((const _Float16*)xin +
                        (size_t)(t0 + xrow) * NDIM + (ks + 2) * KT + xs);
        else      pf = (const char*)((const float*)xin +
                        (size_t)(t0 + xrow) * NDIM + (ks + 2) * KT + xs);
        __builtin_prefetch(pf, 0, 1);
      }
      stage(ks + 1, buf ^ 1);   // overlaps with WMMAs below; waited next iter
    }

    // ---- compute from `buf`: two k-phases of 32 ----
    const _Float16* bx = &lx[buf][0];
    const _Float16* bw = &lw[buf][0];
#pragma unroll
    for (int kh = 0; kh < 2; ++kh) {
      v16h afrag[2];
#pragma unroll
      for (int tt = 0; tt < 2; ++tt) {
        int row = wy * 32 + tt * 16 + llow;
        const _Float16* base = bx + row * LP + kh * 32;
        v8h lo = *(const v8h*)(base + lhalf * 8);
        v8h hi = *(const v8h*)(base + 16 + lhalf * 8);
        afrag[tt] = cat16(lo, hi);
      }
#pragma unroll
      for (int ff = 0; ff < 4; ++ff) {
        int wr = wx * 64 + ff * 16 + llow;
        const _Float16* base = bw + wr * LP + kh * 32;
        v8h lo = *(const v8h*)(base + lhalf * 8);
        v8h hi = *(const v8h*)(base + 16 + lhalf * 8);
        v16h bfrag = cat16(lo, hi);
#pragma unroll
        for (int tt = 0; tt < 2; ++tt)
          acc[tt][ff] = __builtin_amdgcn_wmma_f32_16x16x32_f16(
              false, afrag[tt], false, bfrag, (short)0, acc[tt][ff],
              false, false);
      }
    }
  }

  // ---- epilogue: C/D layout VGPR r -> token r (lanes 0-15) / r+8 (lanes 16-31) ----
#pragma unroll
  for (int tt = 0; tt < 2; ++tt) {
#pragma unroll
    for (int ff = 0; ff < 4; ++ff) {
#pragma unroll
      for (int r = 0; r < 8; ++r) {
        int token = t0 + wy * 32 + tt * 16 + r + lhalf * 8;
        int feat  = m0 + wx * 64 + ff * 16 + llow;
        out[(size_t)token * MDIM + feat] = acc[tt][ff][r] * SCALEV;
      }
    }
  }
}

extern "C" void kernel_launch(void* const* d_in, const int* in_sizes, int n_in,
                              void* d_out, int out_size, void* d_ws, size_t ws_size,
                              hipStream_t stream) {
  const float* x    = (const float*)d_in[0];
  const int*   idxs = (const int*)d_in[1];
  const float* grid = (const float*)d_in[2];
  float*       out  = (float*)d_out;

  dim3 gdim(MDIM / BM, TOKN / BT);  // 64 x 32 blocks
  const size_t need = (size_t)TOKN * NDIM * sizeof(_Float16);  // 64 MB

  if (ws_size >= need) {
    _Float16* xh = (_Float16*)d_ws;
    const int nblk = (int)(((size_t)TOKN * NDIM) / (256 * 8));  // 16384
    xcvt_kernel<<<dim3(nblk), dim3(256), 0, stream>>>(x, xh);
    e8p_gemm_kernel<true><<<gdim, dim3(256), 0, stream>>>(xh, idxs, grid, out);
  } else {
    e8p_gemm_kernel<false><<<gdim, dim3(256), 0, stream>>>(x, idxs, grid, out);
  }
}
